// GewaNet_52802327937491
// MI455X (gfx1250) — matrix operand order repaired and checked by reading, not compiled, not run on hardware
//
#include <hip/hip_runtime.h>
#include <hip/hip_bf16.h>

typedef _Float16 f16;
typedef __attribute__((ext_vector_type(16))) _Float16 v16h;
typedef __attribute__((ext_vector_type(8)))  _Float16 v8h;
typedef __attribute__((ext_vector_type(8)))  float    v8f;

// ---------------------------------------------------------------------------
// WMMA fragment helpers (v_wmma_f32_16x16x32_f16, wave32)
// A tile 16x32 (MxK) 16-bit layout: lanes 0-15 -> row=lane,  K={0..7,16..23}
//                                   lanes16-31 -> row=lane-16,K={8..15,24..31}
// Each half is 8 contiguous f16 -> two 16B vector loads per fragment.
// ---------------------------------------------------------------------------
__device__ inline v16h frag_a_lds(const f16* a, int lda, int lane) {
  int row = lane & 15;
  int kb  = (lane & 16) ? 8 : 0;
  const f16* p = a + row * lda + kb;
  v8h lo = *(const v8h*)p;          // K = kb .. kb+7
  v8h hi = *(const v8h*)(p + 16);   // K = kb+16 .. kb+23
  return __builtin_shufflevector(lo, hi, 0, 1, 2, 3, 4, 5, 6, 7,
                                 8, 9, 10, 11, 12, 13, 14, 15);
}

// B fragments are pre-packed in "fragment order": tile (kt,nt) occupies 512
// contiguous f16; lane l reads its 16 f16 at offset l*16 -> one 32B load.
__device__ inline v16h frag_b_packed(const f16* wtile, int lane) {
  return *(const v16h*)(wtile + lane * 16);
}

__device__ inline v8f wmma_f16(v16h a, v16h b, v8f c) {
  return __builtin_amdgcn_wmma_f32_16x16x32_f16(false, a, false, b, (short)0, c,
                                                false, false);
}

// C/D 16x16 f32 layout: VGPR v: lanes 0-15 -> M=v, lanes 16-31 -> M=v+8; N=lane&15
__device__ inline void store_c_lds(f16* out, int ldo, int row0, int col0, v8f c,
                                   const float* bias, bool relu, int lane) {
  int col = col0 + (lane & 15);
  int rb  = row0 + ((lane & 16) ? 8 : 0);
  float bv = bias[col];
#pragma unroll
  for (int v = 0; v < 8; ++v) {
    float x = c[v] + bv;
    if (relu) x = fmaxf(x, 0.0f);
    out[(rb + v) * ldo + col] = (f16)x;
  }
}

// Per-wave GEMM over an LDS-resident 64xKpad activation slab (Kpad % 32 == 0).
// Wave w computes row-tile w (16 rows) for every 16-col tile of packed W.
__device__ inline void wave_gemm(const f16* albs, int lda, int Kpad,
                                 const f16* wp, int Nf, const float* bias,
                                 f16* outlds, int ldo, bool relu) {
  int lane = threadIdx.x & 31;
  int wave = threadIdx.x >> 5;
  int nkt = Kpad >> 5;
  int ntile = Nf >> 4;
  const f16* arow = albs + wave * 16 * lda;
  for (int ct = 0; ct < ntile; ++ct) {
    const f16* wtile = wp + (size_t)ct * nkt * 512;
    __builtin_prefetch(wtile + nkt * 512, 0, 1);
    v8f acc = {};
    for (int kt = 0; kt < nkt; ++kt) {
      v16h a = frag_a_lds(arow + kt * 32, lda, lane);
      v16h b = frag_b_packed(wtile + kt * 512, lane);
      acc = wmma_f16(a, b, acc);
    }
    store_c_lds(outlds, ldo, wave * 16, ct * 16, acc, bias, relu, lane);
  }
}

// ---------------------------------------------------------------------------
// KNN: one block per query point. pos cached in LDS, 64 iterative arg-mins.
// ---------------------------------------------------------------------------
__global__ __launch_bounds__(256) void knn_kernel(const float* __restrict__ pos,
                                                  int* __restrict__ knn_idx,
                                                  float* __restrict__ knn_d2,
                                                  int N) {
  __shared__ float px[2048], py[2048], pz[2048];
  __shared__ unsigned char taken[2048];
  __shared__ float rmin[256];
  __shared__ int   rarg[256];
  int q = blockIdx.x;
  int b = q / N, n = q % N;
  int tid = threadIdx.x;
  for (int i = tid; i < N; i += 256) {
    px[i] = pos[(size_t)(b * N + i) * 3 + 0];
    py[i] = pos[(size_t)(b * N + i) * 3 + 1];
    pz[i] = pos[(size_t)(b * N + i) * 3 + 2];
    taken[i] = 0;
  }
  __syncthreads();
  float qx = px[n], qy = py[n], qz = pz[n];
  for (int s = 0; s < 64; ++s) {
    float best = 3.0e38f;
    int bestj = 0x7fffffff;
    for (int j = tid; j < N; j += 256) {
      if (taken[j]) continue;
      float dx = px[j] - qx, dy = py[j] - qy, dz = pz[j] - qz;
      float d2 = dx * dx + dy * dy + dz * dz;
      if (d2 < best || (d2 == best && j < bestj)) { best = d2; bestj = j; }
    }
    rmin[tid] = best; rarg[tid] = bestj;
    __syncthreads();
    for (int off = 128; off > 0; off >>= 1) {
      if (tid < off) {
        float om = rmin[tid + off]; int oa = rarg[tid + off];
        if (om < rmin[tid] || (om == rmin[tid] && oa < rarg[tid])) {
          rmin[tid] = om; rarg[tid] = oa;
        }
      }
      __syncthreads();
    }
    if (tid == 0) {
      taken[rarg[0]] = 1;
      knn_idx[(size_t)q * 64 + s] = rarg[0];
      knn_d2[(size_t)q * 64 + s] = rmin[0];
    }
    __syncthreads();
  }
}

// ---------------------------------------------------------------------------
// SA layer: one block (128 thr = 4 waves) per query point.
// Gathered 64 x Kpad feature tile -> 3-layer MLP (WMMA) -> masked max.
// ---------------------------------------------------------------------------
__global__ __launch_bounds__(128) void sa_kernel(
    const float* __restrict__ pos, const int* __restrict__ knn_idx,
    const float* __restrict__ knn_d2, const f16* __restrict__ xin, int cin,
    int kpad0,
    const f16* __restrict__ w0, const float* __restrict__ b0, int n0,
    const f16* __restrict__ w1, const float* __restrict__ b1, int n1,
    const f16* __restrict__ w2, const float* __restrict__ b2, int n2,
    float radius2, f16* __restrict__ xout, int N) {
  __shared__ __align__(32) f16 featbuf[64 * 192];   // K slab (zero padded)
  __shared__ __align__(32) f16 actbuf[64 * 256];    // activation slab
  __shared__ unsigned char validm[64];
  int q = blockIdx.x;
  int b = q / N, n = q % N;
  int tid = threadIdx.x;

  // zero featbuf (vectorized: 64*192 f16 = 1536 uint4)
  for (int i = tid; i < 1536; i += 128) {
    uint4 z; z.x = 0; z.y = 0; z.z = 0; z.w = 0;
    ((uint4*)featbuf)[i] = z;
  }
  __syncthreads();

  float qx = pos[(size_t)(b * N + n) * 3 + 0];
  float qy = pos[(size_t)(b * N + n) * 3 + 1];
  float qz = pos[(size_t)(b * N + n) * 3 + 2];
  for (int r = tid; r < 64; r += 128) {
    int j = knn_idx[(size_t)q * 64 + r];
    validm[r] = (knn_d2[(size_t)q * 64 + r] <= radius2) ? 1 : 0;
    f16* row = featbuf + r * 192;
    if (xin) {  // copy cin (multiple of 8) f16 features, 16B vectors
      const v8h* s = (const v8h*)(xin + (size_t)(b * N + j) * cin);
      v8h* d = (v8h*)row;
      for (int c = 0; c < (cin >> 3); ++c) d[c] = s[c];
    }
    row[cin + 0] = (f16)(pos[(size_t)(b * N + j) * 3 + 0] - qx);
    row[cin + 1] = (f16)(pos[(size_t)(b * N + j) * 3 + 1] - qy);
    row[cin + 2] = (f16)(pos[(size_t)(b * N + j) * 3 + 2] - qz);
  }
  __syncthreads();

  wave_gemm(featbuf, 192, kpad0, w0, n0, b0, actbuf, 256, true);
  __syncthreads();
  wave_gemm(actbuf, 256, n0, w1, n1, b1, featbuf, 192, true);
  __syncthreads();
  wave_gemm(featbuf, 192, n1, w2, n2, b2, actbuf, 256, true);
  __syncthreads();

  for (int c = tid; c < n2; c += 128) {
    float m = -3.0e38f;
    for (int r = 0; r < 64; ++r)
      if (validm[r]) m = fmaxf(m, (float)actbuf[r * 256 + c]);
    xout[(size_t)q * n2 + c] = (f16)m;
  }
}

// ---------------------------------------------------------------------------
// Generic tiled WMMA GEMM: out = relu(A(MxK) * W(KxNf) + bias).
// K already padded (multiple of 64); W in fragment-packed order.
// Block computes a 64x64 tile; optional per-batch column max into `scene`.
// ---------------------------------------------------------------------------
__global__ __launch_bounds__(128) void gemm_kernel(
    const f16* __restrict__ A, int lda, int K,
    const f16* __restrict__ Wp, const float* __restrict__ bias, int Nf,
    f16* __restrict__ out, int ldo, float* __restrict__ scene,
    int rows_per_batch) {
  __shared__ __align__(32) f16 abuf[64 * 64];
  int row0 = blockIdx.x * 64;
  int col0 = blockIdx.y * 64;
  int tid = threadIdx.x, lane = tid & 31, wave = tid >> 5;
  int nkt = K >> 5;
  v8f acc[4];
#pragma unroll
  for (int j = 0; j < 4; ++j)
#pragma unroll
    for (int e = 0; e < 8; ++e) acc[j][e] = 0.0f;

  for (int kk = 0; kk < K; kk += 64) {
    // stage A chunk 64x64 f16 as 512 uint4 (16B) copies
    for (int i = tid; i < 512; i += 128) {
      int r = i >> 3, c = i & 7;
      ((uint4*)abuf)[i] =
          *(const uint4*)(A + (size_t)(row0 + r) * lda + kk + c * 8);
    }
    __syncthreads();
    int ktbase = kk >> 5;
#pragma unroll
    for (int j = 0; j < 4; ++j) {
      const f16* wcol = Wp + ((size_t)((col0 >> 4) + j) * nkt + ktbase) * 512;
      __builtin_prefetch(wcol + 2 * 512, 0, 1);
#pragma unroll
      for (int k2 = 0; k2 < 2; ++k2) {
        v16h a = frag_a_lds(abuf + wave * 16 * 64 + k2 * 32, 64, lane);
        v16h b = frag_b_packed(wcol + k2 * 512, lane);
        acc[j] = wmma_f16(a, b, acc[j]);
      }
    }
    __syncthreads();
  }
#pragma unroll
  for (int j = 0; j < 4; ++j) {
    int col = col0 + j * 16 + (lane & 15);
    int rbase = row0 + wave * 16 + ((lane & 16) ? 8 : 0);
    float bv = bias[col];
#pragma unroll
    for (int v = 0; v < 8; ++v) {
      float x = fmaxf(acc[j][v] + bv, 0.0f);
      if (out) out[(size_t)(rbase + v) * ldo + col] = (f16)x;
      if (scene) {
        int bidx = (rbase + v) / rows_per_batch;
        atomicMax((int*)&scene[(size_t)bidx * Nf + col], __float_as_int(x));
      }
    }
  }
}

// ---------------------------------------------------------------------------
// Weight repack: W (KxN row-major f32) -> fragment-packed f16, zero-padded to
// Kpad. Tile (kt,nt) = 512 contiguous f16; element (l,e): col = nt*16+(l&15),
// k = kt*32 + ((e&8)?16:0) + ((l&16)?8:0) + (e&7). Tiles ordered kt-major within each nt.
// ---------------------------------------------------------------------------
__global__ void repack_w_kernel(const float* __restrict__ src,
                                f16* __restrict__ dst, int K, int N, int Kpad) {
  int i = blockIdx.x * blockDim.x + threadIdx.x;
  if (i >= Kpad * N) return;
  int tile = i >> 9;
  int within = i & 511;
  int l = within >> 4;
  int e = within & 15;
  int nkt = Kpad >> 5;
  int kt = tile % nkt;
  int nt = tile / nkt;
  int col = nt * 16 + (l & 15);
  int k = kt * 32 + ((e & 8) ? 16 : 0) + ((l & 16) ? 8 : 0) + (e & 7);
  dst[i] = (k < K) ? (f16)src[(size_t)k * N + col] : (f16)0.0f;
}

__global__ void zero_f32_kernel(float* __restrict__ p, int n) {
  int i = blockIdx.x * blockDim.x + threadIdx.x;
  if (i < n) p[i] = 0.0f;
}

// feat3 = [x2 (256), pos (3), zeros] padded to width 320
__global__ void build_feat3_kernel(const f16* __restrict__ x2,
                                   const float* __restrict__ pos,
                                   f16* __restrict__ feat3, int total) {
  int i = blockIdx.x * blockDim.x + threadIdx.x;
  int row = i / 320;
  if (row >= total) return;
  int c = i % 320;
  f16 v = (f16)0.0f;
  if (c < 256)      v = x2[(size_t)row * 256 + c];
  else if (c < 259) v = (f16)pos[(size_t)row * 3 + (c - 256)];
  feat3[i] = v;
}

// ---------------------------------------------------------------------------
// Pred MLP + grasp transform (tiny: B rows). fp32 VALU.
// ---------------------------------------------------------------------------
__global__ __launch_bounds__(256) void pred_kernel(
    const f16* __restrict__ x2, const float* __restrict__ scene,
    const int* __restrict__ qidx,
    const float* w0, const float* b0,   // 1280x512
    const float* w1, const float* b1,   // 512x256
    const float* w2, const float* b2,   // 256x9
    float* __restrict__ out, int N) {
  __shared__ float ef[1280];
  __shared__ float h1[512];
  __shared__ float h2[256];
  __shared__ float o9[9];
  int b = blockIdx.x, tid = threadIdx.x;
  int qi = qidx[b];
  for (int i = tid; i < 256; i += 256)
    ef[i] = (float)x2[(size_t)(b * N + qi) * 256 + i];
  for (int i = tid; i < 1024; i += 256) ef[256 + i] = scene[(size_t)b * 1024 + i];
  __syncthreads();
  for (int o = tid; o < 512; o += 256) {
    float s = b0[o];
    for (int k = 0; k < 1280; ++k) s += ef[k] * w0[(size_t)k * 512 + o];
    h1[o] = fmaxf(s, 0.0f);
  }
  __syncthreads();
  for (int o = tid; o < 256; o += 256) {
    float s = b1[o];
    for (int k = 0; k < 512; ++k) s += h1[k] * w1[(size_t)k * 256 + o];
    h2[o] = fmaxf(s, 0.0f);
  }
  __syncthreads();
  if (tid < 9) {
    float s = b2[tid];
    for (int k = 0; k < 256; ++k) s += h2[k] * w2[(size_t)k * 9 + tid];
    o9[tid] = s;  // no activation on last layer
  }
  __syncthreads();
  if (tid == 0) {
    float t0 = o9[0], t1 = o9[1], t2 = o9[2];
    float r1x = o9[3], r1y = o9[4], r1z = o9[5];
    float r2x = o9[6], r2y = o9[7], r2z = o9[8];
    float in1 = rsqrtf(r1x * r1x + r1y * r1y + r1z * r1z);
    r1x *= in1; r1y *= in1; r1z *= in1;
    float d = r1x * r2x + r1y * r2y + r1z * r2z;
    r2x -= d * r1x; r2y -= d * r1y; r2z -= d * r1z;
    float in2 = rsqrtf(r2x * r2x + r2y * r2y + r2z * r2z);
    r2x *= in2; r2y *= in2; r2z *= in2;
    float r3x = r1y * r2z - r1z * r2y;
    float r3y = r1z * r2x - r1x * r2z;
    float r3z = r1x * r2y - r1y * r2x;
    float* O = out + b * 16;
    O[0] = r1x; O[1] = r2x; O[2]  = r3x; O[3]  = t0;
    O[4] = r1y; O[5] = r2y; O[6]  = r3y; O[7]  = t1;
    O[8] = r1z; O[9] = r2z; O[10] = r3z; O[11] = t2;
    O[12] = 0.0f; O[13] = 0.0f; O[14] = 0.0f; O[15] = 1.0f;
  }
}

// ---------------------------------------------------------------------------
// Host launcher
// ---------------------------------------------------------------------------
extern "C" void kernel_launch(void* const* d_in, const int* in_sizes, int n_in,
                              void* d_out, int out_size, void* d_ws,
                              size_t ws_size, hipStream_t stream) {
  (void)n_in; (void)out_size; (void)ws_size;
  const float* pos  = (const float*)d_in[0];
  const int*   qidx = (const int*)d_in[1];
  const int B = in_sizes[1];
  const int N = in_sizes[0] / (3 * B);
  const int BN = B * N;

  // param pointers (dict order: sa1, sa2, sa3, pred; each [w,b]x3)
  const float* sa1_w[3] = {(const float*)d_in[2], (const float*)d_in[4], (const float*)d_in[6]};
  const float* sa1_b[3] = {(const float*)d_in[3], (const float*)d_in[5], (const float*)d_in[7]};
  const float* sa2_w[3] = {(const float*)d_in[8], (const float*)d_in[10], (const float*)d_in[12]};
  const float* sa2_b[3] = {(const float*)d_in[9], (const float*)d_in[11], (const float*)d_in[13]};
  const float* sa3_w[3] = {(const float*)d_in[14], (const float*)d_in[16], (const float*)d_in[18]};
  const float* sa3_b[3] = {(const float*)d_in[15], (const float*)d_in[17], (const float*)d_in[19]};
  const float* pr_w[3]  = {(const float*)d_in[20], (const float*)d_in[22], (const float*)d_in[24]};
  const float* pr_b[3]  = {(const float*)d_in[21], (const float*)d_in[23], (const float*)d_in[25]};

  // K (true), Kpad, Nout per layer
  const int sa1_K[3]  = {3, 64, 64},    sa1_Kp[3] = {32, 64, 64},    sa1_N[3] = {64, 64, 128};
  const int sa2_K[3]  = {131, 128, 128}, sa2_Kp[3] = {160, 128, 128}, sa2_N[3] = {128, 128, 256};
  const int sa3_K[3]  = {259, 256, 512}, sa3_Kp[3] = {320, 256, 512}, sa3_N[3] = {256, 512, 1024};

  // workspace carving
  size_t off = 0;
  auto carve = [&](size_t bytes) {
    void* p = (char*)d_ws + off;
    off += (bytes + 255) & ~(size_t)255;
    return p;
  };
  int*   knn_idx = (int*)carve((size_t)BN * 64 * sizeof(int));
  float* knn_d2  = (float*)carve((size_t)BN * 64 * sizeof(float));
  f16*   x1      = (f16*)carve((size_t)BN * 128 * sizeof(f16));
  f16*   x2      = (f16*)carve((size_t)BN * 256 * sizeof(f16));
  f16*   feat3   = (f16*)carve((size_t)BN * 320 * sizeof(f16));
  f16*   g1      = (f16*)carve((size_t)BN * 256 * sizeof(f16));
  f16*   g2      = (f16*)carve((size_t)BN * 512 * sizeof(f16));
  float* scene   = (float*)carve((size_t)B * 1024 * sizeof(float));
  f16* wp_sa1[3]; f16* wp_sa2[3]; f16* wp_sa3[3];
  for (int l = 0; l < 3; ++l)
    wp_sa1[l] = (f16*)carve((size_t)sa1_Kp[l] * sa1_N[l] * sizeof(f16));
  for (int l = 0; l < 3; ++l)
    wp_sa2[l] = (f16*)carve((size_t)sa2_Kp[l] * sa2_N[l] * sizeof(f16));
  for (int l = 0; l < 3; ++l)
    wp_sa3[l] = (f16*)carve((size_t)sa3_Kp[l] * sa3_N[l] * sizeof(f16));

  // 1. repack weights into fragment-packed f16 (zero-padded K)
  for (int l = 0; l < 3; ++l) {
    int t1 = sa1_Kp[l] * sa1_N[l];
    repack_w_kernel<<<(t1 + 255) / 256, 256, 0, stream>>>(
        sa1_w[l], wp_sa1[l], sa1_K[l], sa1_N[l], sa1_Kp[l]);
    int t2 = sa2_Kp[l] * sa2_N[l];
    repack_w_kernel<<<(t2 + 255) / 256, 256, 0, stream>>>(
        sa2_w[l], wp_sa2[l], sa2_K[l], sa2_N[l], sa2_Kp[l]);
    int t3 = sa3_Kp[l] * sa3_N[l];
    repack_w_kernel<<<(t3 + 255) / 256, 256, 0, stream>>>(
        sa3_w[l], wp_sa3[l], sa3_K[l], sa3_N[l], sa3_Kp[l]);
  }
  zero_f32_kernel<<<(B * 1024 + 255) / 256, 256, 0, stream>>>(scene, B * 1024);

  // 2. knn
  knn_kernel<<<BN, 256, 0, stream>>>(pos, knn_idx, knn_d2, N);

  // 3. sa1: feat = rel(3)->pad32 -> 64 -> 64 -> 128, masked max -> x1
  sa_kernel<<<BN, 128, 0, stream>>>(pos, knn_idx, knn_d2, (const f16*)nullptr, 0,
                                    32,
                                    wp_sa1[0], sa1_b[0], 64,
                                    wp_sa1[1], sa1_b[1], 64,
                                    wp_sa1[2], sa1_b[2], 128,
                                    0.2f * 0.2f, x1, N);

  // 4. sa2: feat = [gather(x1), rel] (131)->pad160 -> 128 -> 128 -> 256 -> x2
  sa_kernel<<<BN, 128, 0, stream>>>(pos, knn_idx, knn_d2, x1, 128,
                                    160,
                                    wp_sa2[0], sa2_b[0], 128,
                                    wp_sa2[1], sa2_b[1], 128,
                                    wp_sa2[2], sa2_b[2], 256,
                                    0.4f * 0.4f, x2, N);

  // 5. sa3: feat3 = [x2, pos] pad320 -> 256 -> 512 -> 1024, max over N -> scene
  {
    int tot = BN * 320;
    build_feat3_kernel<<<(tot + 255) / 256, 256, 0, stream>>>(x2, pos, feat3, BN);
  }
  gemm_kernel<<<dim3(BN / 64, 256 / 64), 128, 0, stream>>>(
      feat3, 320, 320, wp_sa3[0], sa3_b[0], 256, g1, 256, nullptr, N);
  gemm_kernel<<<dim3(BN / 64, 512 / 64), 128, 0, stream>>>(
      g1, 256, 256, wp_sa3[1], sa3_b[1], 512, g2, 512, nullptr, N);
  gemm_kernel<<<dim3(BN / 64, 1024 / 64), 128, 0, stream>>>(
      g2, 512, 512, wp_sa3[2], sa3_b[2], 1024, nullptr, 0, scene, N);

  // 6. pred + transform
  pred_kernel<<<B, 256, 0, stream>>>(x2, scene, qidx,
                                     pr_w[0], pr_b[0], pr_w[1], pr_b[1],
                                     pr_w[2], pr_b[2], (float*)d_out, N);
}
